// GCN_91216515432580
// MI455X (gfx1250) — matrix operand order repaired and checked by reading, not compiled, not run on hardware
//
#include <hip/hip_runtime.h>
#include <hip/hip_bf16.h>
#include <cstdint>
#include <cstddef>

// ---------------- types for WMMA fragments ----------------
typedef __attribute__((ext_vector_type(16))) __bf16 v16bf;
typedef __attribute__((ext_vector_type(8)))  float  v8f;

struct BF16x16Bits { uint4 lo; uint4 hi; };  // 32 bytes = 16 bf16

__device__ __forceinline__ v16bf make_frag16(uint4 lo, uint4 hi) {
    BF16x16Bits u{lo, hi};
    return __builtin_bit_cast(v16bf, u);
}

__device__ __forceinline__ unsigned short f2bf(float f) {
    // round-to-nearest-even f32 -> bf16
    unsigned int u = __float_as_uint(f);
    unsigned int r = u + 0x7FFFu + ((u >> 16) & 1u);
    return (unsigned short)(r >> 16);
}

// ---------------- utility kernels ----------------
__global__ void zero_kernel(float* __restrict__ p, long long n) {
    long long i = (long long)blockIdx.x * blockDim.x + threadIdx.x;
    if (i < n) p[i] = 0.0f;
}

__global__ void deg_kernel(const int* __restrict__ src, const int* __restrict__ dst,
                           float* __restrict__ odeg, float* __restrict__ ideg, int E) {
    int e = blockIdx.x * blockDim.x + threadIdx.x;
    if (e < E) {
        atomicAdd(&odeg[src[e]], 1.0f);
        atomicAdd(&ideg[dst[e]], 1.0f);
    }
}

__global__ void norm_kernel(const float* __restrict__ odeg, const float* __restrict__ ideg,
                            float* __restrict__ onrm, float* __restrict__ inrm, int N) {
    int i = blockIdx.x * blockDim.x + threadIdx.x;
    if (i < N) {
        onrm[i] = rsqrtf(fmaxf(odeg[i], 1.0f));
        inrm[i] = rsqrtf(fmaxf(ideg[i], 1.0f));
    }
}

// W is [K x nout] row-major f32; write Wt as [noutPad x K] bf16 (transposed, zero-padded)
__global__ void prepw_kernel(const float* __restrict__ W, unsigned short* __restrict__ Wt,
                             int K, int nout, int noutPad) {
    int t = blockIdx.x * blockDim.x + threadIdx.x;
    if (t >= noutPad * K) return;
    int n = t / K;
    int k = t - n * K;
    float v = (n < nout) ? W[(size_t)k * nout + n] : 0.0f;
    Wt[t] = f2bf(v);
}

// Xb[i] = bf16(h[i] * out_norm[i/128]); input width fixed at 128
__global__ void scale_cvt_kernel(const float* __restrict__ h, const float* __restrict__ onrm,
                                 unsigned short* __restrict__ Xb, long long total) {
    long long i = (long long)blockIdx.x * blockDim.x + threadIdx.x;
    if (i < total) Xb[i] = f2bf(h[i] * onrm[i >> 7]);
}

// ---------------- WMMA GEMM: Y[N x ncols] = Xb[N x 128] @ Wt^T ----------------
// block = 256 threads = 8 waves; wave w handles column tile w; blockIdx.x = 16-row tile.
// A tile (16x128 bf16, 4KB) staged through LDS once per block, shared by all waves.
__global__ void gemm_bf16_wmma(const unsigned short* __restrict__ Xb,
                               const unsigned short* __restrict__ Wt,
                               float* __restrict__ Y,
                               int Nrows, int ncols, int ntilesCol) {
    constexpr int K = 128;
    __shared__ unsigned short Atile[16 * K];   // 4 KB

    const int rowTile = blockIdx.x;
    const int tid = threadIdx.x;

    // cooperative A-tile load: 2048 ushorts, 8 per thread (one b128 each)
    {
        int elem = tid * 8;                    // 0..2040
        int r = elem >> 7;                     // row within tile
        int kk = elem & 127;
        int gr = rowTile * 16 + r;
        if (gr >= Nrows) gr = Nrows - 1;       // clamp (duplicate row, never OOB)
        *reinterpret_cast<uint4*>(&Atile[elem]) =
            *reinterpret_cast<const uint4*>(Xb + (size_t)gr * K + kk);
    }
    __syncthreads();

    const int wave = tid >> 5;
    if (wave >= ntilesCol) return;             // uniform per wave: EXEC stays all-1s
    const int lane = tid & 31;
    const int half = lane >> 4;                // 0 | 1
    const int l15  = lane & 15;

    const unsigned short* arow = &Atile[l15 * K];
    const unsigned short* brow = Wt + (size_t)(wave * 16 + l15) * K;

    v8f acc = {0.f, 0.f, 0.f, 0.f, 0.f, 0.f, 0.f, 0.f};

#pragma unroll
    for (int kb = 0; kb < K; kb += 32) {
        // A 16x32 bf16 frag: lane = row, 16 contiguous K at kb + half*16
        v16bf A = make_frag16(
            *reinterpret_cast<const uint4*>(arow + kb + half * 16),
            *reinterpret_cast<const uint4*>(arow + kb + half * 16 + 8));
        // B 32x16 bf16 frag: lane = column, 16 contiguous K at kb + half*16
        v16bf B = make_frag16(
            *reinterpret_cast<const uint4*>(brow + kb + half * 16),
            *reinterpret_cast<const uint4*>(brow + kb + half * 16 + 8));
        acc = __builtin_amdgcn_wmma_f32_16x16x32_bf16(
            /*neg_a=*/false, A, /*neg_b=*/false, B,
            /*c_mod=*/(short)0, acc, /*reuse_a=*/false, /*reuse_b=*/false);
    }

    // D layout: VGPR r -> row (r + half*8), lane&15 -> col
    const int colBase = wave * 16 + l15;
    float* yp = Y + (size_t)(rowTile * 16 + half * 8) * ncols + colBase;
    if (rowTile * 16 + 16 <= Nrows) {
        // fast path: full tile, unguarded stores
#pragma unroll
        for (int r = 0; r < 8; ++r) {
            yp[(size_t)r * ncols] = acc[r];
        }
    } else {
        // last partial tile only
#pragma unroll
        for (int r = 0; r < 8; ++r) {
            int m = rowTile * 16 + r + half * 8;
            if (m < Nrows) yp[(size_t)r * ncols] = acc[r];
        }
    }
}

// ---------------- edge scatter: agg[dst] += Y[src] ----------------
__global__ void scatter_kernel(const float* __restrict__ Y, const int* __restrict__ src,
                               const int* __restrict__ dst, float* __restrict__ agg,
                               int E, int stride, int chunks) {
    long long t = (long long)blockIdx.x * blockDim.x + threadIdx.x;
    long long total = (long long)E * chunks;
    if (t >= total) return;
    int e = (int)((unsigned long long)t / (unsigned)chunks);
    int c = (int)(t - (long long)e * chunks);
    const float4 v = *reinterpret_cast<const float4*>(Y + (size_t)src[e] * stride + c * 4);
    float* a = agg + (size_t)dst[e] * stride + c * 4;
    atomicAdd(a + 0, v.x);
    atomicAdd(a + 1, v.y);
    atomicAdd(a + 2, v.z);
    atomicAdd(a + 3, v.w);
}

// out[node, f] = act(agg[node*strideIn + f] * in_norm[node] + bias[f])
__global__ void finalize_kernel(const float* __restrict__ agg, const float* __restrict__ inrm,
                                const float* __restrict__ bias, float* __restrict__ out,
                                int N, int strideIn, int dout, int doRelu) {
    long long i = (long long)blockIdx.x * blockDim.x + threadIdx.x;
    long long total = (long long)N * dout;
    if (i >= total) return;
    int node = (int)((unsigned long long)i / (unsigned)dout);
    int f = (int)(i - (long long)node * dout);
    float v = agg[(size_t)node * strideIn + f] * inrm[node] + bias[f];
    out[i] = doRelu ? fmaxf(v, 0.0f) : v;
}

// ---------------- host launcher ----------------
extern "C" void kernel_launch(void* const* d_in, const int* in_sizes, int n_in,
                              void* d_out, int out_size, void* d_ws, size_t ws_size,
                              hipStream_t stream) {
    (void)n_in; (void)out_size; (void)ws_size;

    const float* feat = (const float*)d_in[0];
    const float* W1   = (const float*)d_in[1];
    const float* b1   = (const float*)d_in[2];
    const float* W2   = (const float*)d_in[3];
    const float* b2   = (const float*)d_in[4];
    const float* W3   = (const float*)d_in[5];
    const float* b3   = (const float*)d_in[6];
    const int*   src  = (const int*)d_in[7];
    const int*   dst  = (const int*)d_in[8];

    const int D   = 128;                 // IN_FEAT == HIDDEN == 128
    const int NC  = 40;                  // classes
    const int NCP = 48;                  // padded to 3 WMMA col tiles
    const int N   = in_sizes[0] / D;     // 50000
    const int E   = in_sizes[7];         // 800000

    // workspace carve-out (256B aligned slices)
    char* ws = (char*)d_ws;
    size_t off = 0;
    auto take = [&](size_t bytes) -> char* {
        char* p = ws + off;
        off = (off + bytes + 255) & ~(size_t)255;
        return p;
    };
    float*          odeg = (float*)take((size_t)N * 4);
    float*          ideg = (float*)take((size_t)N * 4);
    float*          onrm = (float*)take((size_t)N * 4);
    float*          inrm = (float*)take((size_t)N * 4);
    unsigned short* Wt1  = (unsigned short*)take((size_t)D * D * 2);
    unsigned short* Wt2  = (unsigned short*)take((size_t)D * D * 2);
    unsigned short* Wt3  = (unsigned short*)take((size_t)NCP * D * 2);
    unsigned short* Xb   = (unsigned short*)take((size_t)N * D * 2);
    float*          Ybuf = (float*)take((size_t)N * D * 4);   // also reused as hidden h
    float*          agg  = (float*)take((size_t)N * D * 4);

    const dim3 blk(256);
    auto g1 = [](long long n) { return dim3((unsigned)((n + 255) / 256)); };
    const int rowTiles = (N + 15) / 16;

    // degrees + norms
    zero_kernel<<<g1(N), blk, 0, stream>>>(odeg, N);
    zero_kernel<<<g1(N), blk, 0, stream>>>(ideg, N);
    deg_kernel<<<g1(E), blk, 0, stream>>>(src, dst, odeg, ideg, E);
    norm_kernel<<<g1(N), blk, 0, stream>>>(odeg, ideg, onrm, inrm, N);

    // weight prep (transpose + bf16 + pad)
    prepw_kernel<<<g1((long long)D * D), blk, 0, stream>>>(W1, Wt1, D, D, D);
    prepw_kernel<<<g1((long long)D * D), blk, 0, stream>>>(W2, Wt2, D, D, D);
    prepw_kernel<<<g1((long long)NCP * D), blk, 0, stream>>>(W3, Wt3, D, NC, NCP);

    auto layer = [&](const float* hin, const unsigned short* Wt, const float* bias,
                     int ncols, int ncolsPad, float* outbuf, int relu) {
        scale_cvt_kernel<<<g1((long long)N * D), blk, 0, stream>>>(hin, onrm, Xb, (long long)N * D);
        gemm_bf16_wmma<<<dim3(rowTiles), blk, 0, stream>>>(Xb, Wt, Ybuf, N, ncolsPad, ncolsPad / 16);
        zero_kernel<<<g1((long long)N * ncolsPad), blk, 0, stream>>>(agg, (long long)N * ncolsPad);
        scatter_kernel<<<g1((long long)E * (ncols / 4)), blk, 0, stream>>>(Ybuf, src, dst, agg, E, ncolsPad, ncols / 4);
        finalize_kernel<<<g1((long long)N * ncols), blk, 0, stream>>>(agg, inrm, bias, outbuf, N, ncolsPad, ncols, relu);
    };

    layer(feat, Wt1, b1, D, D, Ybuf, 1);                 // h1 = relu(gconv(feat))
    layer(Ybuf, Wt2, b2, D, D, Ybuf, 1);                 // h2 = relu(gconv(h1))
    layer(Ybuf, Wt3, b3, NC, NCP, (float*)d_out, 0);     // logits = gconv(h2)
}
